// ConditionGraphRoutingNetwork_5411658793122
// MI455X (gfx1250) — compile-verified, matrix operand order checked by reading, not run on hardware
//
#include <hip/hip_runtime.h>
#include <hip/hip_bf16.h>
#include <cstdint>
#include <cstddef>

// ---------------- problem constants ----------------
#define F_DIM   128
#define H_DIM   128
#define KEXP    8
#define T_WIN   32
#define NNODE   512
#define NPE     63              // 2*T - 1
#define NEG_SLOPE 0.01f

typedef __attribute__((ext_vector_type(8)))  float  v8f;
typedef __attribute__((ext_vector_type(16))) __bf16 v16bf;
typedef __attribute__((ext_vector_type(4)))  unsigned int u32x4;
typedef __attribute__((ext_vector_type(8)))  int i32x8;
typedef __attribute__((ext_vector_type(4)))  int i32x4;

union Frag {
  v16bf v;
  __bf16 h[16];
  unsigned short u[16];
  unsigned int   w[8];
};

__device__ __forceinline__ float leaky(float x) { return x > 0.f ? x : NEG_SLOPE * x; }

// float -> bf16 via hardware convert (v_cvt_*bf16*), RNE
__device__ __forceinline__ unsigned short f2bf(float x) {
  union { __bf16 b; unsigned short u; } c;
  c.b = (__bf16)x;
  return c.u;
}

// wave32 16-bit A/B operand K-offset for VGPR p and lane-half
__device__ __forceinline__ int kmap(int p, int half) {
  return (p < 4) ? (8 * half + 2 * p) : (16 + 8 * half + 2 * (p - 4));
}

// ---------------- workspace layout (float offsets) ----------------
#define OFF_TABS 0
#define OFF_TABE 512
#define OFF_SI   1024
#define OFF_SJ   5120
#define OFF_ATTI 9216
#define OFF_ATTJ 13312
#define OFF_UI   17408
#define OFF_UJ   18432
#define OFF_CI   19456
#define OFF_CJ   19464
#define OFF_WVT  19968   // bf16 WV_t[8][128][512], 16B aligned

// =====================================================================
// TDM: load B panel WV_t[0..8)[0..128)[j0..j0+32) into LDS (64KB tile)
// 3D tile: dim0=j (32, contig), dim1=h (128, stride 512), dim2=k (8,
// stride 65536), data_size=2B. Issued by one wave; tracked by TENSORcnt.
// =====================================================================
__device__ __forceinline__ void tdm_load_b_panel(const unsigned short* gptr,
                                                 unsigned lds_addr) {
  const unsigned long long ga = (unsigned long long)(uintptr_t)gptr;
  u32x4 g0;
  g0[0] = 1u;                                           // count=1, user mode
  g0[1] = lds_addr;                                     // LDS byte address
  g0[2] = (unsigned)(ga & 0xFFFFFFFFu);                 // global_addr[31:0]
  g0[3] = (unsigned)((ga >> 32) & 0x01FFFFFFu) | (2u << 30); // addr[56:32]|type=2
  i32x8 g1;
  g1[0] = 0x10000;                 // data_size=1 (2 bytes)
  g1[1] = (int)(512u << 16);       // tensor_dim0[15:0]=512 at bits[63:48]
  g1[2] = (int)(128u << 16);       // tensor_dim0[31:16]=0 | tensor_dim1[15:0]=128
  g1[3] = (int)(32u  << 16);       // tensor_dim1[31:16]=0 | tile_dim0=32
  g1[4] = (int)(128u | (8u << 16));// tile_dim1=128 | tile_dim2=8
  g1[5] = 512;                     // tensor_dim0_stride[31:0]=512
  g1[6] = 0;                       // dim0_stride[47:32]=0 | dim1_stride[15:0]=0
  g1[7] = 1;                       // dim1_stride[47:16]=1 -> 65536
  i32x4 g2;
  g2[0] = 8;                       // tensor_dim2 = 8 (k)
  g2[1] = 1;                       // tensor_dim3 (unused, tile_dim3=0)
  g2[2] = 65536;                   // tensor_dim2_stride[31:0] = 128*512
  g2[3] = 0;                       // dim2_stride[47:32]=0 | tile_dim3=0
  i32x4 g3 = {0, 0, 0, 0};
#if defined(__clang_major__) && (__clang_major__ >= 23)
  i32x8 g4 = {0, 0, 0, 0, 0, 0, 0, 0};
  __builtin_amdgcn_tensor_load_to_lds(g0, g1, g2, g3, g4, 0);
#else
  __builtin_amdgcn_tensor_load_to_lds(g0, g1, g2, g3, 0);
#endif
}

// =====================================================================
// Kernel 1: positional-encoding tables  tab_s[p][k], tab_e[p][k]
// grid = 63, block = 128
// =====================================================================
__global__ void k_pe(const float* __restrict__ Wa, const float* __restrict__ a,
                     float* __restrict__ tabs, float* __restrict__ tabe) {
  __shared__ float pe[F_DIM];
  const int p = blockIdx.x;
  const int d = threadIdx.x;
  const float pos = (float)p - (float)(T_WIN - 1);
  float val;
  if (d < 64) {
    float div = __powf(10000.f, (2.f * (float)d) / (float)F_DIM);
    val = __sinf(pos / div);
  } else {
    int dd = d - 64;
    float div = __powf(10000.f, (2.f * (float)dd) / (float)F_DIM);
    val = __cosf(pos / div);
  }
  pe[d] = val;
  __syncthreads();
  if (d < 16) {
    const int k = d & 7;
    const float* wrow = (d < 8) ? (Wa + k * 384 + 256) : (a + k * 384 + 256);
    float s = 0.f;
    for (int f = 0; f < F_DIM; ++f) s += pe[f] * wrow[f];
    if (d < 8) tabs[p * 8 + k] = s; else tabe[p * 8 + k] = s;
  }
}

// =====================================================================
// Kernel 2: u_i[k][f] = Wk[k] @ a_i[k],  u_j,  c_i[k]=bk[k].a_i[k], c_j
// grid = 8, block = 128
// =====================================================================
__global__ void k_u(const float* __restrict__ Wk, const float* __restrict__ a,
                    const float* __restrict__ bk,
                    float* __restrict__ ui, float* __restrict__ uj,
                    float* __restrict__ ci, float* __restrict__ cj) {
  const int k = blockIdx.x;
  const int f = threadIdx.x;
  const float* wrow = Wk + ((size_t)k * F_DIM + f) * H_DIM;
  const float* ai = a + k * 384;
  const float* aj = ai + H_DIM;
  float si = 0.f, sj = 0.f;
  for (int h = 0; h < H_DIM; ++h) { si += wrow[h] * ai[h]; sj += wrow[h] * aj[h]; }
  ui[k * F_DIM + f] = si;
  uj[k * F_DIM + f] = sj;
  if (f == 0) { float c = 0.f; for (int h = 0; h < H_DIM; ++h) c += bk[k * H_DIM + h] * ai[h]; ci[k] = c; }
  if (f == 1) { float c = 0.f; for (int h = 0; h < H_DIM; ++h) c += bk[k * H_DIM + h] * aj[h]; cj[k] = c; }
}

// =====================================================================
// Kernel 3: per-node dots  s_i[n][k], s_j[n][k], att_i[k][n], att_j[k][n]
// grid = 512, block = 128
// =====================================================================
__global__ void k_node(const float* __restrict__ V, const float* __restrict__ Wa,
                       const float* __restrict__ ui, const float* __restrict__ uj,
                       const float* __restrict__ ci, const float* __restrict__ cj,
                       float* __restrict__ si, float* __restrict__ sj,
                       float* __restrict__ atti, float* __restrict__ attj) {
  __shared__ float v[F_DIM];
  const int n = blockIdx.x;
  v[threadIdx.x] = V[(size_t)n * F_DIM + threadIdx.x];
  __syncthreads();
  const int t = threadIdx.x;
  if (t < 32) {
    const int k = t & 7;
    const int sel = t >> 3;
    const float* w;
    float s = 0.f;
    if (sel == 0)      w = Wa + k * 384;
    else if (sel == 1) w = Wa + k * 384 + F_DIM;
    else if (sel == 2) { w = ui + k * F_DIM; s = ci[k]; }
    else               { w = uj + k * F_DIM; s = cj[k]; }
    for (int f = 0; f < F_DIM; ++f) s += v[f] * w[f];
    if (sel == 0)      si[n * 8 + k] = s;
    else if (sel == 1) sj[n * 8 + k] = s;
    else if (sel == 2) atti[k * NNODE + n] = s;
    else               attj[k * NNODE + n] = s;
  }
}

// =====================================================================
// Kernel 4: WV = V @ Wk[k] + bk[k], stored transposed bf16 WV_t[k][h][n]
// 2048 tiles (k,nt,ht) of 16x16; 8 waves/block -> grid 256, block 256
// =====================================================================
__global__ void k_wv(const float* __restrict__ V, const float* __restrict__ Wk,
                     const float* __restrict__ bk, unsigned short* __restrict__ WVt) {
  const int tid  = threadIdx.x;
  const int lane = tid & 31;
  const int wave = tid >> 5;
  const int m16  = lane & 15;
  const int half = lane >> 4;
  const int tile = blockIdx.x * 8 + wave;       // k*256 + nt*8 + ht
  const int ht = tile & 7;
  const int nt = (tile >> 3) & 31;
  const int k  = tile >> 8;
  const int n0 = nt * 16, h0 = ht * 16;

  const float* Vrow = V + (size_t)(n0 + m16) * F_DIM;              // A: row m
  const float* Wkc  = Wk + (size_t)k * F_DIM * H_DIM + (h0 + m16); // B: column n(=h)

  v8f acc = {};
  #pragma unroll
  for (int f0 = 0; f0 < F_DIM; f0 += 32) {
    Frag A, B;
    #pragma unroll
    for (int p = 0; p < 8; ++p) {
      const int kk = kmap(p, half);
      A.h[2 * p]     = (__bf16)Vrow[f0 + kk];
      A.h[2 * p + 1] = (__bf16)Vrow[f0 + kk + 1];
      B.h[2 * p]     = (__bf16)Wkc[(size_t)(f0 + kk) * H_DIM];
      B.h[2 * p + 1] = (__bf16)Wkc[(size_t)(f0 + kk + 1) * H_DIM];
    }
    acc = __builtin_amdgcn_wmma_f32_16x16x32_bf16(false, A.v, false, B.v,
                                                  (short)0, acc, false, false);
  }
  const float bkv = bk[k * H_DIM + h0 + m16];
  Frag o;
  #pragma unroll
  for (int r = 0; r < 8; ++r) o.u[r] = f2bf(acc[r] + bkv);
  // D: lane holds column h=h0+m16, rows n = n0 + 8*half + r  (contiguous in WV_t)
  const size_t base = ((size_t)(k * H_DIM + h0 + m16)) * NNODE + (size_t)n0 + 8 * half;
  uint4 st; st.x = o.w[0]; st.y = o.w[1]; st.z = o.w[2]; st.w = o.w[3];
  *reinterpret_cast<uint4*>(WVt + base) = st;
}

// =====================================================================
// Kernel 5: fused routing + dual softmax + messages GEMM + output leaky
// grid = 32 (16 i-rows each), block = 256 (8 waves; wave w owns h-tile w)
// dynamic LDS: 64KB B panel, filled by TDM tensor_load_to_lds
// =====================================================================
__global__ void __launch_bounds__(256)
k_fused(const int* __restrict__ Dt, const float* __restrict__ Eg,
        const float* __restrict__ g_tabs, const float* __restrict__ g_tabe,
        const float* __restrict__ g_si, const float* __restrict__ g_sj,
        const float* __restrict__ g_atti, const float* __restrict__ g_attj,
        const unsigned short* __restrict__ WVt, float* __restrict__ out) {
  extern __shared__ char smem[];
  unsigned short* Bl = (unsigned short*)smem;        // [8][128][32] bf16 = 64KB

  __shared__ float tab_s_l[NPE * 8];
  __shared__ float tab_e_l[NPE * 8];
  __shared__ float atti_l[KEXP * 16];
  __shared__ float rowmax_l[KEXP * 16];
  __shared__ float rowsum_l[KEXP * 16];
  __shared__ unsigned short wtile[KEXP * 16 * 32];   // bf16 w[k][i_local][jj]

  const int tid  = threadIdx.x;
  const int lane = tid & 31;
  const int wave = tid >> 5;
  const int m16  = lane & 15;
  const int half = lane >> 4;
  const int i_base = blockIdx.x * 16;
  const int hcol = wave * 16 + m16;
  const unsigned bl_lds = (unsigned)(uintptr_t)Bl;   // LDS byte address (low 32b)

  for (int t = tid; t < NPE * 8; t += 256) { tab_s_l[t] = g_tabs[t]; tab_e_l[t] = g_tabe[t]; }
  if (tid < KEXP * 16) {
    const int k = tid >> 4, il = tid & 15;
    atti_l[tid] = g_atti[k * NNODE + i_base + il];
  }
  __syncthreads();

  // ---- pass 1: per-(k,i) online softmax stats over all j ----
  #pragma unroll
  for (int ii = 0; ii < 2; ++ii) {
    const int il = wave * 2 + ii;
    const int i  = i_base + il;
    float ai[KEXP];
    #pragma unroll
    for (int k = 0; k < KEXP; ++k) ai[k] = atti_l[k * 16 + il];
    float m[KEXP], s[KEXP];
    #pragma unroll
    for (int k = 0; k < KEXP; ++k) { m[k] = -3.4e38f; s[k] = 0.f; }
    for (int j = lane; j < NNODE; j += 32) {
      const int idxv = (Dt[(size_t)i * NNODE + j] + T_WIN - 1) * 8;
      #pragma unroll
      for (int k = 0; k < KEXP; ++k) {
        const float e = leaky(ai[k] + g_attj[k * NNODE + j] + tab_e_l[idxv + k]);
        if (e > m[k]) { s[k] = s[k] * __expf(m[k] - e) + 1.f; m[k] = e; }
        else          { s[k] += __expf(e - m[k]); }
      }
    }
    #pragma unroll
    for (int k = 0; k < KEXP; ++k) {
      float mk = m[k], sk = s[k];
      for (int off = 16; off > 0; off >>= 1) {
        const float om = __shfl_xor(mk, off, 32);
        const float os = __shfl_xor(sk, off, 32);
        const float nm = fmaxf(mk, om);
        sk = sk * __expf(mk - nm) + os * __expf(om - nm);
        mk = nm;
      }
      if (lane == 0) { rowmax_l[k * 16 + il] = mk; rowsum_l[k * 16 + il] = sk; }
    }
  }
  __syncthreads();

  // ---- pass 2: TDM-prefetch B panel, build w tiles, WMMA ----
  v8f acc = {};
  for (int j0 = 0; j0 < NNODE; j0 += 32) {
    // wave 0 kicks off the async 64KB B-panel DMA for this j0 slice;
    // it overlaps with the whole gating/softmax VALU pass below.
    if (wave == 0 && lane == 0) tdm_load_b_panel(WVt + j0, bl_lds);

    for (int q = tid; q < 512; q += 256) {
      const int il = q >> 5, jj = q & 31;
      const int i = i_base + il, j = j0 + jj;
      const int idxv = (Dt[(size_t)i * NNODE + j] + T_WIN - 1) * 8;
      const float Eij = Eg[(size_t)i * NNODE + j];
      const float esq = Eij * Eij;
      const float* sig = g_si + (size_t)i * 8;
      const float* sjg = g_sj + (size_t)j * 8;
      float sc[KEXP];
      #pragma unroll
      for (int k = 0; k < KEXP; ++k) sc[k] = leaky(sig[k] + sjg[k] + tab_s_l[idxv + k]);
      int k1 = 0;
      #pragma unroll
      for (int k = 1; k < KEXP; ++k) if (sc[k] > sc[k1]) k1 = k;
      int k2 = -1;
      #pragma unroll
      for (int k = 0; k < KEXP; ++k) if (k != k1 && (k2 < 0 || sc[k] > sc[k2])) k2 = k;
      const float e1 = __expf(sc[k1]);
      const float e2 = __expf(sc[k2]);
      const float inv = 1.f / (e1 + e2 + 1e-8f);
      #pragma unroll
      for (int k = 0; k < KEXP; ++k) {
        const float e = leaky(atti_l[k * 16 + il] + g_attj[k * NNODE + j] + tab_e_l[idxv + k]);
        const float att = __expf(e - rowmax_l[k * 16 + il]) / rowsum_l[k * 16 + il];
        const float g = (k == k1) ? e1 * inv : ((k == k2) ? e2 * inv : 0.f);
        wtile[(k * 16 + il) * 32 + jj] = f2bf(att * g * esq);
      }
    }
    if (wave == 0) __builtin_amdgcn_s_wait_tensorcnt(0);  // DMA done (wave0 only)
    __syncthreads();                                      // publish Bl + wtile

    #pragma unroll
    for (int k = 0; k < KEXP; ++k) {
      Frag A, B;
      #pragma unroll
      for (int p = 0; p < 8; ++p) {
        const int kk = kmap(p, half);
        A.w[p] = *reinterpret_cast<const unsigned int*>(&wtile[(k * 16 + m16) * 32 + kk]);
        B.w[p] = *reinterpret_cast<const unsigned int*>(&Bl[(k * H_DIM + hcol) * 32 + kk]);
      }
      acc = __builtin_amdgcn_wmma_f32_16x16x32_bf16(false, A.v, false, B.v,
                                                    (short)0, acc, false, false);
    }
    __syncthreads();   // Bl/wtile consumed before next iteration overwrites
  }

  // ---- write 16x16 output tile: out[i][h] = leaky(acc) ----
  #pragma unroll
  for (int r = 0; r < 8; ++r) {
    const int i = i_base + r + 8 * half;
    out[(size_t)i * H_DIM + hcol] = leaky(acc[r]);
  }
}

// =====================================================================
// launch
// =====================================================================
extern "C" void kernel_launch(void* const* d_in, const int* in_sizes, int n_in,
                              void* d_out, int out_size, void* d_ws, size_t ws_size,
                              hipStream_t stream) {
  const float* V   = (const float*)d_in[0];
  const int*   Dt  = (const int*)  d_in[1];
  const float* Eg  = (const float*)d_in[2];
  const float* Wa  = (const float*)d_in[3];
  const float* Wk  = (const float*)d_in[4];
  const float* bk  = (const float*)d_in[5];
  const float* a   = (const float*)d_in[6];
  float* out = (float*)d_out;

  float* ws = (float*)d_ws;
  float* tabs = ws + OFF_TABS;
  float* tabe = ws + OFF_TABE;
  float* si   = ws + OFF_SI;
  float* sj   = ws + OFF_SJ;
  float* atti = ws + OFF_ATTI;
  float* attj = ws + OFF_ATTJ;
  float* ui   = ws + OFF_UI;
  float* uj   = ws + OFF_UJ;
  float* ci   = ws + OFF_CI;
  float* cj   = ws + OFF_CJ;
  unsigned short* WVt = (unsigned short*)(ws + OFF_WVT);

  k_pe  <<<NPE,   128, 0, stream>>>(Wa, a, tabs, tabe);
  k_u   <<<KEXP,  128, 0, stream>>>(Wk, a, bk, ui, uj, ci, cj);
  k_node<<<NNODE, 128, 0, stream>>>(V, Wa, ui, uj, ci, cj, si, sj, atti, attj);
  k_wv  <<<256,   256, 0, stream>>>(V, Wk, bk, WVt);
  k_fused<<<NNODE / 16, 256, 65536, stream>>>(Dt, Eg, tabs, tabe, si, sj,
                                              atti, attj, WVt, out);
}
